// EncoderBlock_77489799954932
// MI455X (gfx1250) — matrix operand order repaired and checked by reading, not compiled, not run on hardware
//
#include <hip/hip_runtime.h>
#include <hip/hip_bf16.h>

#define Bb   2
#define Tt   2048
#define Dd   1024
#define Hh   16
#define HDd  64
#define DFF  4096
#define Mm   (Bb*Tt)
#define EPSf 1e-6f

typedef _Float16 h16;
typedef __attribute__((ext_vector_type(16))) _Float16 v16h;
typedef __attribute__((ext_vector_type(8)))  _Float16 h8v;
typedef __attribute__((ext_vector_type(8)))  float    v8f;

// ---------------------------------------------------------------- helpers
__device__ __forceinline__ v8f wmma_f16(v16h a, v16h b, v8f c) {
  // D = A(16x32 f16) * B(32x16 f16) + C(16x16 f32)
  return __builtin_amdgcn_wmma_f32_16x16x32_f16(false, a, false, b,
                                                (short)0, c, false, false);
}

// K-index permutation inside a 32-wide chunk so that each lane's 16 A-matrix
// halves are CONTIGUOUS: positions [0..15] = K 0-7,16-23 (lanes 0-15),
// positions [16..31] = K 8-15,24-31 (lanes 16-31).
__device__ __forceinline__ int permk(int w) {
  const int m[4] = {0, 16, 8, 24};
  return m[(w >> 3) & 3] + (w & 7);
}

// gfx1250 async global->LDS copy (16 bytes/lane), tracked by ASYNCcnt.
// VDST = LDS byte address (addrspace(3) offset), VADDR = 64-bit global addr.
__device__ __forceinline__ void async_load_b128(void* lds_dst, const void* gsrc) {
  unsigned lo = (unsigned)(size_t)(__attribute__((address_space(3))) const void*)lds_dst;
  asm volatile("global_load_async_to_lds_b128 %0, %1, off"
               :: "v"(lo), "v"((unsigned long long)(size_t)gsrc)
               : "memory");
}
__device__ __forceinline__ void wait_async0() {
  asm volatile("s_wait_asynccnt 0x0" ::: "memory");
}

// ---------------------------------------------------------------- rmsnorm
__global__ __launch_bounds__(256) void rmsnorm_kernel(const float* __restrict__ x,
                                                      const float* __restrict__ scale,
                                                      h16* __restrict__ out) {
  __shared__ float red[8];
  __shared__ float srs;
  const int row  = blockIdx.x;
  const int tid  = threadIdx.x;
  const int lane = tid & 31, wid = tid >> 5;
  const float* xr = x + (size_t)row * Dd;
  float ss = 0.f;
  for (int i = tid; i < Dd; i += 256) { float v = xr[i]; ss += v * v; }
#pragma unroll
  for (int m = 1; m < 32; m <<= 1) ss += __shfl_xor(ss, m, 32);
  if (lane == 0) red[wid] = ss;
  __syncthreads();
  if (tid == 0) {
    float tot = 0.f;
#pragma unroll
    for (int i = 0; i < 8; ++i) tot += red[i];
    srs = rsqrtf(tot * (1.0f / Dd) + EPSf);
  }
  __syncthreads();
  const float rs = srs;
  for (int i = tid; i < Dd; i += 256)
    out[(size_t)row * Dd + i] = (h16)(xr[i] * rs * scale[i]);
}

// ---------------------------------------------------------------- f32 -> f16
__global__ __launch_bounds__(256) void f32_to_f16_kernel(const float* __restrict__ in,
                                                         h16* __restrict__ out, int n) {
  int i = blockIdx.x * 256 + threadIdx.x;
  if (i < n) out[i] = (h16)in[i];
}

// ---------------------------------------------------------------- WMMA GEMM
// C[M,N] = A[M,K] * B[K,N]; EPI==0 -> store f16; EPI==1 -> f32 = resid + acc
template <int EPI>
__global__ __launch_bounds__(256) void gemm_wmma(const h16* __restrict__ A,
                                                 const h16* __restrict__ Bw,
                                                 void* __restrict__ outp,
                                                 const float* __restrict__ resid,
                                                 int M, int N, int K) {
  __shared__ alignas(64) h16 lds_a[128 * 32];   // [row][permuted k]
  __shared__ alignas(64) h16 lds_b[128 * 32];   // [n][k] transposed
  const int tid  = threadIdx.x;
  const int wid  = tid >> 5, lane = tid & 31;
  const int bm   = blockIdx.x * 128;
  const int bn   = blockIdx.y * 128;
  const int wm   = (wid >> 1) * 32;   // wave row offset (0..96)
  const int wn   = (wid & 1) * 64;    // wave col offset (0,64)
  const int mrow = lane & 15, l2 = lane >> 4;

  v8f acc[2][4];
#pragma unroll
  for (int i = 0; i < 2; ++i)
#pragma unroll
    for (int j = 0; j < 4; ++j)
#pragma unroll
      for (int r = 0; r < 8; ++r) acc[i][j][r] = 0.f;

  const int arow = tid >> 1, ak = (tid & 1) * 16;   // A-tile loader mapping
  const int pa   = ak >> 1;                         // permuted store base (0 or 8)
  const int kr   = tid >> 3, nc = (tid & 7) * 16;   // B-tile loader mapping

  for (int kb = 0; kb < K; kb += 32) {
    if (kb) __syncthreads();
    const h16* ga = A  + (size_t)(bm + arow) * K + kb + ak;
    const h16* gb = Bw + (size_t)(kb + kr) * N + bn + nc;
    // stage A: async HBM->LDS copies; permuted layout moves 16B groups as units
    async_load_b128(lds_a + arow * 32 + pa,      ga);
    async_load_b128(lds_a + arow * 32 + pa + 16, ga + 8);
    // stage B transposed -> [n][k] (element transpose: VGPR path)
    {
      h8v b0 = *(const h8v*)gb;
      h8v b1 = *(const h8v*)(gb + 8);
#pragma unroll
      for (int j = 0; j < 8; ++j) {
        lds_b[(nc + j) * 32 + permk(kr)]     = b0[j];
        lds_b[(nc + 8 + j) * 32 + permk(kr)] = b1[j];
      }
    }
    // prefetch next k-chunk tiles into cache while WMMAs run
    if (kb + 32 < K) {
      __builtin_prefetch(ga + 32, 0, 0);
      __builtin_prefetch(gb + (size_t)32 * N, 0, 0);
    }
    wait_async0();
    __syncthreads();

    v16h af[2], bf[4];
#pragma unroll
    for (int i = 0; i < 2; ++i)
      af[i] = *(const v16h*)(lds_a + (wm + i * 16 + mrow) * 32 + l2 * 16);
#pragma unroll
    for (int j = 0; j < 4; ++j)
      bf[j] = *(const v16h*)(lds_b + (wn + j * 16 + mrow) * 32 + l2 * 16);
#pragma unroll
    for (int i = 0; i < 2; ++i)
#pragma unroll
      for (int j = 0; j < 4; ++j)
        acc[i][j] = wmma_f16(af[i], bf[j], acc[i][j]);
  }

  // epilogue: C layout = VGPR r, row = r + 8*(lane/16), col = lane&15
#pragma unroll
  for (int i = 0; i < 2; ++i)
#pragma unroll
    for (int j = 0; j < 4; ++j)
#pragma unroll
      for (int r = 0; r < 8; ++r) {
        const int row = bm + wm + i * 16 + r + 8 * l2;
        const int col = bn + wn + j * 16 + mrow;
        const size_t idx = (size_t)row * N + col;
        if (EPI == 0) ((h16*)outp)[idx] = (h16)acc[i][j][r];
        else          ((float*)outp)[idx] = resid[idx] + acc[i][j][r];
      }
}

// ---------------------------------------------------------------- RoPE + head transpose
// in: [B,T,H*HD] f16 ; out: [B,H,T,HD] f16.
// PERM=1 writes each 32-wide half in WMMA-A permuted order (for Q fragments).
template <int PERM>
__global__ __launch_bounds__(256) void rope_kernel(const h16* __restrict__ in,
                                                   const float* __restrict__ cosT,
                                                   const float* __restrict__ sinT,
                                                   h16* __restrict__ out) {
  const int idx = blockIdx.x * 256 + threadIdx.x;   // B*T*H*32 total
  const int i = idx & 31;
  const int h = (idx >> 5) & (Hh - 1);
  const int t = (idx >> 9) & (Tt - 1);
  const int b = idx >> 20;
  const float c = cosT[t * 32 + i], s = sinT[t * 32 + i];
  const size_t bi = (size_t)(b * Tt + t) * Dd + h * HDd;
  const float x1 = (float)in[bi + i];
  const float x2 = (float)in[bi + 32 + i];
  const size_t bo = (size_t)((b * Hh + h) * Tt + t) * HDd;
  const int p = PERM ? permk(i) : i;
  out[bo + p]      = (h16)(x1 * c - x2 * s);
  out[bo + 32 + p] = (h16)(x2 * c + x1 * s);
}

// V: [B,T,H*HD] -> [B,H,HD,T]  (d-major so PV B-fragments are contiguous)
__global__ __launch_bounds__(256) void vtrans_kernel(const h16* __restrict__ in,
                                                     h16* __restrict__ out) {
  const int idx = blockIdx.x * 256 + threadIdx.x;   // B*T*H*HD total
  const int d = idx & (HDd - 1);
  const int h = (idx >> 6) & (Hh - 1);
  const int t = (idx >> 10) & (Tt - 1);
  const int b = idx >> 21;
  out[((size_t)(b * Hh + h) * HDd + d) * Tt + t] =
      in[(size_t)(b * Tt + t) * Dd + h * HDd + d];
}

// ---------------------------------------------------------------- flash attention
// q: [B,H,T,HD] (permuted-K), k: [B,H,T,HD], v: [B,H,HD,T]; out: [B,T,H*HD].
// Each wave owns 16 query rows -> no cross-wave softmax traffic.
__global__ __launch_bounds__(256) void attn_kernel(const h16* __restrict__ qh,
                                                   const h16* __restrict__ kh,
                                                   const h16* __restrict__ vt,
                                                   h16* __restrict__ out) {
  __shared__ alignas(64) h16 p_lds[8][16 * 32];
  const int tid = threadIdx.x;
  const int wid = tid >> 5, lane = tid & 31;
  const int mrow = lane & 15, l2 = lane >> 4;
  const int bh = blockIdx.y;            // b*H + h
  const int b = bh / Hh, h = bh % Hh;
  const int q0 = blockIdx.x * 128 + wid * 16;
  const h16* Qb = qh + (size_t)bh * Tt * HDd;
  const h16* Kb = kh + (size_t)bh * Tt * HDd;
  const h16* Vt = vt + (size_t)bh * HDd * Tt;

  // Q A-fragments: contiguous thanks to permuted layout
  v16h aq[2];
#pragma unroll
  for (int c = 0; c < 2; ++c)
    aq[c] = *(const v16h*)(Qb + (size_t)(q0 + mrow) * HDd + c * 32 + l2 * 16);

  float mrun[8], lrun[8];
  v8f acco[4];
#pragma unroll
  for (int r = 0; r < 8; ++r) { mrun[r] = -3.0e38f; lrun[r] = 0.f; }
#pragma unroll
  for (int n = 0; n < 4; ++n)
#pragma unroll
    for (int r = 0; r < 8; ++r) acco[n][r] = 0.f;

  const int kend = q0 + 16;             // causal: keys < kend
  for (int kb = 0; kb < kend; kb += 32) {
    // ---- scores: two 16x16 tiles over 32 keys, contraction over HD=64
    v8f s[2];
#pragma unroll
    for (int t2 = 0; t2 < 2; ++t2) {
#pragma unroll
      for (int r = 0; r < 8; ++r) s[t2][r] = 0.f;
#pragma unroll
      for (int c = 0; c < 2; ++c) {
        v16h bk = *(const v16h*)(Kb + (size_t)(kb + t2 * 16 + mrow) * HDd +
                                 c * 32 + l2 * 16);
        s[t2] = wmma_f16(aq[c], bk, s[t2]);
      }
    }
    // ---- scale + causal mask + row max
    float mcur[8];
#pragma unroll
    for (int r = 0; r < 8; ++r) mcur[r] = -3.0e38f;
#pragma unroll
    for (int t2 = 0; t2 < 2; ++t2) {
      const int key = kb + t2 * 16 + mrow;
#pragma unroll
      for (int r = 0; r < 8; ++r) {
        float v = s[t2][r] * 0.125f;            // 1/sqrt(HD)
        const int qrow = q0 + r + 8 * l2;
        if (key > qrow) v = -3.0e38f;
        s[t2][r] = v;
        mcur[r] = fmaxf(mcur[r], v);
      }
    }
#pragma unroll
    for (int m = 1; m < 16; m <<= 1)
#pragma unroll
      for (int r = 0; r < 8; ++r) mcur[r] = fmaxf(mcur[r], __shfl_xor(mcur[r], m, 32));

    float alpha[8], lsum[8];
#pragma unroll
    for (int r = 0; r < 8; ++r) {
      const float mn = fmaxf(mrun[r], mcur[r]);
      alpha[r] = __expf(mrun[r] - mn);
      mrun[r] = mn;
    }
#pragma unroll
    for (int t2 = 0; t2 < 2; ++t2)
#pragma unroll
      for (int r = 0; r < 8; ++r) s[t2][r] = __expf(s[t2][r] - mrun[r]);
#pragma unroll
    for (int r = 0; r < 8; ++r) lsum[r] = s[0][r] + s[1][r];
#pragma unroll
    for (int m = 1; m < 16; m <<= 1)
#pragma unroll
      for (int r = 0; r < 8; ++r) lsum[r] += __shfl_xor(lsum[r], m, 32);
#pragma unroll
    for (int r = 0; r < 8; ++r) lrun[r] = lrun[r] * alpha[r] + lsum[r];
#pragma unroll
    for (int n = 0; n < 4; ++n)
#pragma unroll
      for (int r = 0; r < 8; ++r) acco[n][r] *= alpha[r];

    // ---- reshape P (C layout) -> contiguous A layout via per-wave LDS tile
    h16* pl = p_lds[wid];
#pragma unroll
    for (int t2 = 0; t2 < 2; ++t2) {
      const int pcol = t2 * 16 + mrow;
#pragma unroll
      for (int r = 0; r < 8; ++r)
        pl[(r + 8 * l2) * 32 + pcol] = (h16)s[t2][r];
    }
    v16h ap = *(const v16h*)(p_lds[wid] + mrow * 32 + l2 * 16);

    // ---- O += P(16x32) * V(32xHD); V transposed -> contiguous B-fragments
#pragma unroll
    for (int n = 0; n < 4; ++n) {
      v16h bv = *(const v16h*)(Vt + (size_t)(n * 16 + mrow) * Tt + kb + l2 * 16);
      acco[n] = wmma_f16(ap, bv, acco[n]);
    }
  }

  // ---- normalize + write out as [B,T,H*HD]
  float rl[8];
#pragma unroll
  for (int r = 0; r < 8; ++r) rl[r] = 1.0f / lrun[r];
#pragma unroll
  for (int n = 0; n < 4; ++n)
#pragma unroll
    for (int r = 0; r < 8; ++r) {
      const int qrow = q0 + r + 8 * l2;
      const int d = n * 16 + mrow;
      out[(size_t)(b * Tt + qrow) * Dd + h * HDd + d] = (h16)(acco[n][r] * rl[r]);
    }
}

// ---------------------------------------------------------------- silu(g)*u
__global__ __launch_bounds__(256) void silu_mul_kernel(const h16* __restrict__ g,
                                                       const h16* __restrict__ u,
                                                       h16* __restrict__ out, int n) {
  int i = blockIdx.x * 256 + threadIdx.x;
  if (i < n) {
    const float x = (float)g[i];
    const float s = x / (1.0f + __expf(-x));
    out[i] = (h16)(s * (float)u[i]);
  }
}

// ---------------------------------------------------------------- launcher
extern "C" void kernel_launch(void* const* d_in, const int* in_sizes, int n_in,
                              void* d_out, int out_size, void* d_ws, size_t ws_size,
                              hipStream_t stream) {
  const float* x    = (const float*)d_in[0];
  // d_in[1] = mask (causal, computed analytically)
  const float* cosT = (const float*)d_in[2];
  const float* sinT = (const float*)d_in[3];
  const float* ln1  = (const float*)d_in[4];
  const float* wq   = (const float*)d_in[5];
  const float* wk   = (const float*)d_in[6];
  const float* wv   = (const float*)d_in[7];
  const float* wo   = (const float*)d_in[8];
  const float* ln2  = (const float*)d_in[9];
  const float* wg   = (const float*)d_in[10];
  const float* wu   = (const float*)d_in[11];
  const float* wd   = (const float*)d_in[12];
  float* out = (float*)d_out;

  char* w = (char*)d_ws;
  size_t off = 0;
  auto takeh = [&](size_t elems) -> h16* {
    h16* p = (h16*)(w + off);
    off = (off + elems * sizeof(h16) + 255) & ~(size_t)255;
    return p;
  };
  h16* h1    = takeh((size_t)Mm * Dd);
  h16* wq16  = takeh((size_t)Dd * Dd);
  h16* wk16  = takeh((size_t)Dd * Dd);
  h16* wv16  = takeh((size_t)Dd * Dd);
  h16* wo16  = takeh((size_t)Dd * Dd);
  h16* wg16  = takeh((size_t)Dd * DFF);
  h16* wu16  = takeh((size_t)Dd * DFF);
  h16* wd16  = takeh((size_t)DFF * Dd);
  h16* q16   = takeh((size_t)Mm * Dd);
  h16* k16   = takeh((size_t)Mm * Dd);
  h16* v16   = takeh((size_t)Mm * Dd);
  h16* qhp   = takeh((size_t)Mm * Dd);
  h16* khp   = takeh((size_t)Mm * Dd);
  h16* vtp   = takeh((size_t)Mm * Dd);
  h16* att16 = takeh((size_t)Mm * Dd);
  float* x1  = (float*)(w + off);
  off = (off + (size_t)Mm * Dd * sizeof(float) + 255) & ~(size_t)255;
  h16* h2    = takeh((size_t)Mm * Dd);
  h16* g16   = takeh((size_t)Mm * DFF);
  h16* u16   = takeh((size_t)Mm * DFF);
  h16* ff16  = takeh((size_t)Mm * DFF);

  auto conv = [&](const float* s, h16* d, int n) {
    f32_to_f16_kernel<<<(n + 255) / 256, 256, 0, stream>>>(s, d, n);
  };
  conv(wq, wq16, Dd * Dd);
  conv(wk, wk16, Dd * Dd);
  conv(wv, wv16, Dd * Dd);
  conv(wo, wo16, Dd * Dd);
  conv(wg, wg16, Dd * DFF);
  conv(wu, wu16, Dd * DFF);
  conv(wd, wd16, DFF * Dd);

  // h = rmsnorm(x) -> f16
  rmsnorm_kernel<<<Mm, 256, 0, stream>>>(x, ln1, h1);

  // QKV projections
  dim3 gD(Mm / 128, Dd / 128);
  gemm_wmma<0><<<gD, 256, 0, stream>>>(h1, wq16, q16, nullptr, Mm, Dd, Dd);
  gemm_wmma<0><<<gD, 256, 0, stream>>>(h1, wk16, k16, nullptr, Mm, Dd, Dd);
  gemm_wmma<0><<<gD, 256, 0, stream>>>(h1, wv16, v16, nullptr, Mm, Dd, Dd);

  // RoPE + head-major transpose (Q in permuted-K layout), V -> [B,H,HD,T]
  const int nrope = Bb * Tt * Hh * 32;
  rope_kernel<1><<<nrope / 256, 256, 0, stream>>>(q16, cosT, sinT, qhp);
  rope_kernel<1><<<nrope / 256, 256, 0, stream>>>(k16, cosT, sinT, khp);
  vtrans_kernel<<<(Mm * Dd) / 256, 256, 0, stream>>>(v16, vtp);

  // flash attention
  attn_kernel<<<dim3(Tt / 128, Bb * Hh), 256, 0, stream>>>(qhp, khp, vtp, att16);

  // x1 = x + attn @ wo   (f32)
  gemm_wmma<1><<<gD, 256, 0, stream>>>(att16, wo16, x1, x, Mm, Dd, Dd);

  // h2 = rmsnorm(x1) -> f16
  rmsnorm_kernel<<<Mm, 256, 0, stream>>>(x1, ln2, h2);

  // MLP
  dim3 gF(Mm / 128, DFF / 128);
  gemm_wmma<0><<<gF, 256, 0, stream>>>(h2, wg16, g16, nullptr, Mm, DFF, Dd);
  gemm_wmma<0><<<gF, 256, 0, stream>>>(h2, wu16, u16, nullptr, Mm, DFF, Dd);
  silu_mul_kernel<<<(Mm * DFF) / 256, 256, 0, stream>>>(g16, u16, ff16, Mm * DFF);

  // out = x1 + ff @ wd   (f32)
  gemm_wmma<1><<<gD, 256, 0, stream>>>(ff16, wd16, out, x1, Mm, Dd, DFF);
}
